// GraphAttentionEmbedding_44882408243776
// MI455X (gfx1250) — matrix operand order repaired and checked by reading, not compiled, not run on hardware
//
#include <hip/hip_runtime.h>

#define NN 50000
#define NE 800000
#define DDIN 256
#define DDOUT 256
#define NHEAD 4
#define CHD 64
#define EDD 128

typedef __bf16 bf16;
typedef __attribute__((ext_vector_type(16))) __bf16 v16bf;
typedef __attribute__((ext_vector_type(8)))  float  v8f;

static __device__ __forceinline__ bf16 f2bf(float f) {
    union { float f; unsigned u; } x; x.f = f;
    unsigned r = x.u + 0x7FFFu + ((x.u >> 16) & 1u);
    union { unsigned short s; bf16 b; } y; y.s = (unsigned short)(r >> 16);
    return y.b;
}
static __device__ __forceinline__ float bf2f(bf16 b) {
    union { unsigned short s; bf16 b; } y; y.b = b;
    union { unsigned u; float f; } x; x.u = ((unsigned)y.s) << 16;
    return x.f;
}

// ---------------- elementwise conversion / zero ----------------
__global__ void cvt_f32_to_bf16_k(const float* __restrict__ in, bf16* __restrict__ out, long n) {
    long i = (long)blockIdx.x * blockDim.x + threadIdx.x;
    long stride = (long)gridDim.x * blockDim.x;
    for (; i < n; i += stride) out[i] = f2bf(in[i]);
}
__global__ void zero_f32_k(float* __restrict__ p, long n) {
    long i = (long)blockIdx.x * blockDim.x + threadIdx.x;
    long stride = (long)gridDim.x * blockDim.x;
    for (; i < n; i += stride) p[i] = 0.0f;
}

// ---------------- weight packing into WMMA B-fragment order ----------------
// B is [K,256] row-major fp32.  Fragment (tk, tn) covers K rows tk*32..+31,
// cols tn*16..+15.  Per-lane layout (this code's contiguous k order, same as the A loads): n = lane&15,
// k = tk*32 + (lane>>4)*16 + i, i = 0..15, stored contiguously (16 bf16/lane).
__global__ void pack_w_bf16_k(const float* __restrict__ W, bf16* __restrict__ Wp, int K) {
    const int NC = 256;
    int idx = blockIdx.x * blockDim.x + threadIdx.x;
    int total = (K >> 5) * (NC >> 4) * 32;
    if (idx >= total) return;
    int lane = idx & 31;
    int frag = idx >> 5;
    int tn = frag & 15;
    int tk = frag >> 4;
    int n  = tn * 16 + (lane & 15);
    int kb = tk * 32 + (lane >> 4) * 16;
    bf16* dst = Wp + (size_t)idx * 16;
#pragma unroll
    for (int i = 0; i < 16; ++i) dst[i] = f2bf(W[(size_t)(kb + i) * NC + n]);
}

// ---------------- bf16 WMMA GEMM:  C[M,256] = A[M,K] @ B[K,256] + bias ----------------
// grid.x = M/16 (M multiple of 16); 128 threads = 4 waves, wave w covers cols w*64..+63
union FragAB { v16bf v; uint4 q[2]; };
union FragC  { v8f v; float f[8]; };

template <bool OUT_BF16>
__global__ __launch_bounds__(128) void gemm_bf16_wmma_k(
    const bf16* __restrict__ A, const bf16* __restrict__ Bp,
    const float* __restrict__ bias, void* __restrict__ Cout, int K) {
    const int NC = 256;
    const int lane  = threadIdx.x & 31;
    const int wave  = threadIdx.x >> 5;
    const int rowB  = blockIdx.x * 16;
    const int m     = lane & 15;
    const int khalf = lane >> 4;

    const bf16* arow = A + (size_t)(rowB + m) * K + khalf * 16;

    FragC c[4];
#pragma unroll
    for (int t = 0; t < 4; ++t)
#pragma unroll
        for (int i = 0; i < 8; ++i) c[t].f[i] = 0.0f;

    const int nKT = K >> 5;
    for (int kt = 0; kt < nKT; ++kt) {
        FragAB a;
        const uint4* ap = (const uint4*)(arow + kt * 32);
        a.q[0] = ap[0];   // K = kt*32 + khalf*16 .. +7
        a.q[1] = ap[1];   // K = kt*32 + khalf*16 + 8 .. +15
#pragma unroll
        for (int t = 0; t < 4; ++t) {
            FragAB b;
            const uint4* bp = (const uint4*)(Bp + (size_t)(kt * 16 + wave * 4 + t) * 512 + lane * 16);
            b.q[0] = bp[0];
            b.q[1] = bp[1];
            c[t].v = __builtin_amdgcn_wmma_f32_16x16x32_bf16(
                false, a.v, false, b.v, (short)0, c[t].v, false, false);
        }
    }

#pragma unroll
    for (int t = 0; t < 4; ++t) {
        int n = wave * 64 + t * 16 + (lane & 15);
        float bs = bias[n];
#pragma unroll
        for (int i = 0; i < 8; ++i) {
            int r = rowB + i + khalf * 8;
            float val = c[t].f[i] + bs;
            if (OUT_BF16) ((bf16*)Cout)[(size_t)r * NC + n] = f2bf(val);
            else          ((float*)Cout)[(size_t)r * NC + n] = val;
        }
    }
}

// ---------------- attention logits + segment max ----------------
// one wave per edge; lane covers 8 contiguous channels (head = lane>>3)
__global__ __launch_bounds__(256) void edge_alpha_k(
    const int* __restrict__ ei, const float* __restrict__ Q, const float* __restrict__ Kf,
    const bf16* __restrict__ Ebf, float* __restrict__ alpha, unsigned* __restrict__ amax_bits) {
    int e = blockIdx.x * 8 + (threadIdx.x >> 5);
    if (e >= NE) return;
    int lane = threadIdx.x & 31;
    int src = ei[e];
    int dst = ei[NE + e];
    const float* q  = Q   + (size_t)dst * DDOUT + lane * 8;
    const float* k  = Kf  + (size_t)src * DDOUT + lane * 8;
    const bf16*  ep = Ebf + (size_t)e   * DDOUT + lane * 8;
    float p = 0.f;
#pragma unroll
    for (int j = 0; j < 8; ++j) p += q[j] * (k[j] + bf2f(ep[j]));
    p += __shfl_xor(p, 1, 32);
    p += __shfl_xor(p, 2, 32);
    p += __shfl_xor(p, 4, 32);
    if ((lane & 7) == 0) {
        int h = lane >> 3;
        float a = p * 0.125f;                 // / sqrt(64)
        alpha[(size_t)e * NHEAD + h] = a;
        unsigned bits = __float_as_uint(a);
        unsigned enc = (bits & 0x80000000u) ? ~bits : (bits | 0x80000000u);
        atomicMax(&amax_bits[(size_t)dst * NHEAD + h], enc);
    }
}

__global__ void amax_finalize_k(unsigned* __restrict__ amax, int n) {
    int i = blockIdx.x * blockDim.x + threadIdx.x;
    if (i >= n) return;
    unsigned u = amax[i];
    unsigned bits = (u & 0x80000000u) ? (u & 0x7FFFFFFFu) : ~u;
    float f = __uint_as_float(bits);
    if (!(f > -3.0e38f && f < 3.0e38f)) f = 0.0f;   // non-finite (untouched) -> 0
    ((float*)amax)[i] = f;
}

__global__ void exp_denom_k(const int* __restrict__ ei, float* __restrict__ alpha,
                            const float* __restrict__ amaxf, float* __restrict__ denom) {
    long i = (long)blockIdx.x * blockDim.x + threadIdx.x;
    if (i >= (long)NE * NHEAD) return;
    int e = (int)(i >> 2);
    int h = (int)(i & 3);
    int dst = ei[NE + e];
    float ex = __expf(alpha[i] - amaxf[(size_t)dst * NHEAD + h]);
    alpha[i] = ex;
    atomicAdd(&denom[(size_t)dst * NHEAD + h], ex);
}

// ---------------- weighted message scatter ----------------
__global__ __launch_bounds__(256) void scatter_msg_k(
    const int* __restrict__ ei, const float* __restrict__ Vf, const bf16* __restrict__ Ebf,
    const float* __restrict__ alpha, const float* __restrict__ denom, float* __restrict__ agg) {
    int e = blockIdx.x * 8 + (threadIdx.x >> 5);
    if (e >= NE) return;
    int lane = threadIdx.x & 31;
    int src = ei[e];
    int dst = ei[NE + e];
    int h = lane >> 3;
    float w = alpha[(size_t)e * NHEAD + h] / (denom[(size_t)dst * NHEAD + h] + 1e-16f);
    const float* v  = Vf  + (size_t)src * DDOUT + lane * 8;
    const bf16*  ep = Ebf + (size_t)e   * DDOUT + lane * 8;
    float* o = agg + (size_t)dst * DDOUT + lane * 8;
#pragma unroll
    for (int j = 0; j < 8; ++j) atomicAdd(&o[j], (v[j] + bf2f(ep[j])) * w);
}

// ---------------- epilogues ----------------
__global__ void epilogue_relu_bf16_k(const float* __restrict__ agg, const float* __restrict__ S,
                                     bf16* __restrict__ hb, long n) {
    long i = (long)blockIdx.x * blockDim.x + threadIdx.x;
    long stride = (long)gridDim.x * blockDim.x;
    for (; i < n; i += stride) {
        float v = agg[i] + S[i];
        hb[i] = f2bf(v > 0.f ? v : 0.f);
    }
}
__global__ void epilogue_add_k(const float* __restrict__ agg, const float* __restrict__ S,
                               float* __restrict__ out, long n) {
    long i = (long)blockIdx.x * blockDim.x + threadIdx.x;
    long stride = (long)gridDim.x * blockDim.x;
    for (; i < n; i += stride) out[i] = agg[i] + S[i];
}

extern "C" void kernel_launch(void* const* d_in, const int* in_sizes, int n_in,
                              void* d_out, int out_size, void* d_ws, size_t ws_size,
                              hipStream_t stream) {
    (void)in_sizes; (void)n_in; (void)out_size; (void)ws_size;

    const float* x  = (const float*)d_in[0];
    const int*   ei = (const int*)d_in[1];
    const float* ea = (const float*)d_in[2];
    const float* Wsrc[10]; const float* Bias[10]; int Kdim[10];
    for (int l = 0; l < 2; ++l)
        for (int j = 0; j < 5; ++j) {
            Wsrc[l * 5 + j] = (const float*)d_in[3 + l * 10 + j * 2];
            Bias[l * 5 + j] = (const float*)d_in[4 + l * 10 + j * 2];
            Kdim[l * 5 + j] = (j == 3) ? EDD : 256;
        }

    // workspace layout
    size_t off = 0;
    auto take = [&](size_t bytes) -> char* {
        char* r = (char*)d_ws + off;
        off += (bytes + 255) & ~(size_t)255;
        return r;
    };
    bf16* xb  = (bf16*)take((size_t)NN * DDIN * 2);
    bf16* hb  = (bf16*)take((size_t)NN * DDOUT * 2);
    bf16* eab = (bf16*)take((size_t)NE * EDD * 2);
    bf16* Wp[10];
    for (int i = 0; i < 10; ++i) Wp[i] = (bf16*)take((size_t)Kdim[i] * 256 * 2);
    float* Qf = (float*)take((size_t)NN * DDOUT * 4);
    float* Kf = (float*)take((size_t)NN * DDOUT * 4);
    float* Vf = (float*)take((size_t)NN * DDOUT * 4);
    float* Sf = (float*)take((size_t)NN * DDOUT * 4);
    bf16*  Ebf = (bf16*)take((size_t)NE * DDOUT * 2);
    float* alpha = (float*)take((size_t)NE * NHEAD * 4);
    unsigned* amax = (unsigned*)take((size_t)NN * NHEAD * 4);
    float* denom = (float*)take((size_t)NN * NHEAD * 4);
    float* agg  = (float*)take((size_t)NN * DDOUT * 4);

    // --- convert inputs to bf16 / pack weights ---
    cvt_f32_to_bf16_k<<<2048, 256, 0, stream>>>(x, xb, (long)NN * DDIN);
    cvt_f32_to_bf16_k<<<4096, 256, 0, stream>>>(ea, eab, (long)NE * EDD);
    for (int i = 0; i < 10; ++i) {
        int tot = (Kdim[i] >> 5) * 16 * 32;
        pack_w_bf16_k<<<(tot + 255) / 256, 256, 0, stream>>>(Wsrc[i], Wp[i], Kdim[i]);
    }

    for (int layer = 0; layer < 2; ++layer) {
        const bf16* Ain = (layer == 0) ? xb : hb;
        const int wb = layer * 5;

        zero_f32_k<<<256, 256, 0, stream>>>((float*)amax, (long)NN * NHEAD);
        zero_f32_k<<<256, 256, 0, stream>>>(denom, (long)NN * NHEAD);
        zero_f32_k<<<4096, 256, 0, stream>>>(agg, (long)NN * DDOUT);

        // node GEMMs (M = 50000 = 3125 * 16)
        gemm_bf16_wmma_k<false><<<NN / 16, 128, 0, stream>>>(Ain, Wp[wb + 0], Bias[wb + 0], Qf, 256);
        gemm_bf16_wmma_k<false><<<NN / 16, 128, 0, stream>>>(Ain, Wp[wb + 1], Bias[wb + 1], Kf, 256);
        gemm_bf16_wmma_k<false><<<NN / 16, 128, 0, stream>>>(Ain, Wp[wb + 2], Bias[wb + 2], Vf, 256);
        gemm_bf16_wmma_k<false><<<NN / 16, 128, 0, stream>>>(Ain, Wp[wb + 4], Bias[wb + 4], Sf, 256);
        // edge GEMM (M = 800000 = 50000 * 16, K = 128), bf16 output
        gemm_bf16_wmma_k<true><<<NE / 16, 128, 0, stream>>>(eab, Wp[wb + 3], Bias[wb + 3], Ebf, EDD);

        edge_alpha_k<<<NE / 8, 256, 0, stream>>>(ei, Qf, Kf, Ebf, alpha, amax);
        amax_finalize_k<<<(NN * NHEAD + 255) / 256, 256, 0, stream>>>(amax, NN * NHEAD);
        exp_denom_k<<<(NE * NHEAD + 255) / 256, 256, 0, stream>>>(ei, alpha, (const float*)amax, denom);
        scatter_msg_k<<<NE / 8, 256, 0, stream>>>(ei, Vf, Ebf, alpha, denom, agg);

        if (layer == 0)
            epilogue_relu_bf16_k<<<4096, 256, 0, stream>>>(agg, Sf, hb, (long)NN * DDOUT);
        else
            epilogue_add_k<<<4096, 256, 0, stream>>>(agg, Sf, (float*)d_out, (long)NN * DDOUT);
    }
}
